// SwinV2UNet256Generator_10436770529358
// MI455X (gfx1250) — compile-verified, matrix operand order checked by reading, not run on hardware
//
#include <hip/hip_runtime.h>
#include <hip/hip_bf16.h>

// SwinV2 windowed cosine attention, fused per-window kernel for gfx1250.
// All matrix math on v_wmma_f32_16x16x32_bf16 (f32 accumulate).
// Tile loops use exact per-wave trip counts so EXEC is statically all-ones
// at every WMMA (ISA 7.12 requirement) and loop control stays scalar.

#define NH       3
#define NW       1024          // (256/8)^2 windows per image
#define N_TOK    64
#define C_DIM    96
#define QKV_OUT  288

typedef __attribute__((ext_vector_type(16))) __bf16 v16bf;
typedef __attribute__((ext_vector_type(8)))  float  v8f;

// Workspace byte offsets
#define WS_WQKV   0                      // bf16 [288*96]   = 55296 B
#define WS_WPROJ  55296                  // bf16 [96*96]    = 18432 B
#define WS_QKVB   73728                  // f32  [288]      =  1152 B
#define WS_BIAS16 74880                  // f32  [3*64*64]  = 49152 B (ends 124032)

// Load a 16x32 bf16 A/B fragment from a row-major matrix (ld elements/row).
// Lane layout per CDNA5 ISA 7.12.2: lanes 0-15 rows r0..r0+15 with K-halves
// [0..7]+[16..23]; lanes 16-31 same rows, K-halves [8..15]+[24..31].
__device__ __forceinline__ v16bf frag_ld(const __bf16* base, int ld, int r0,
                                         int k0, int lane) {
  const __bf16* p = base + (r0 + (lane & 15)) * ld + k0 + ((lane >> 4) << 3);
  v16bf f;
#pragma unroll
  for (int i = 0; i < 4; ++i) {
    f[2 * i]     = p[2 * i];
    f[2 * i + 1] = p[2 * i + 1];
    f[8 + 2 * i] = p[16 + 2 * i];
    f[9 + 2 * i] = p[16 + 2 * i + 1];
  }
  return f;
}

__device__ __forceinline__ v8f wmma_bf16(v16bf a, v16bf b, v8f c) {
  return __builtin_amdgcn_wmma_f32_16x16x32_bf16(false, a, false, b,
                                                 (short)0, c, false, false);
}

// ---------------------------------------------------------------------------
// Prep: weight conversion to bf16, fused qkv bias, CPB-MLP bias table.
// ---------------------------------------------------------------------------
__global__ __launch_bounds__(256) void swin_prep_kernel(
    const float* __restrict__ qkv_w, const float* __restrict__ proj_w,
    const float* __restrict__ q_bias, const float* __restrict__ v_bias,
    const float* __restrict__ cpb_w1, const float* __restrict__ cpb_b1,
    const float* __restrict__ cpb_w2, const float* __restrict__ coords,
    const int* __restrict__ rel_index, void* __restrict__ wsv) {
  __bf16* w_qkv  = (__bf16*)((char*)wsv + WS_WQKV);
  __bf16* w_proj = (__bf16*)((char*)wsv + WS_WPROJ);
  float*  qkvb   = (float*)((char*)wsv + WS_QKVB);
  float*  bias16 = (float*)((char*)wsv + WS_BIAS16);
  const int tid = threadIdx.x;

  for (int i = tid; i < QKV_OUT * C_DIM; i += 256) w_qkv[i] = (__bf16)qkv_w[i];
  for (int i = tid; i < C_DIM * C_DIM; i += 256)  w_proj[i] = (__bf16)proj_w[i];
  for (int i = tid; i < QKV_OUT; i += 256) {
    float b = 0.0f;
    if (i < C_DIM) b = q_bias[i];
    else if (i >= 2 * C_DIM) b = v_bias[i - 2 * C_DIM];
    qkvb[i] = b;
  }

  // CPB MLP: (15*15, 2) -> relu(512) -> (NH,)
  __shared__ float tbl[225 * NH];
  for (int r = tid; r < 225; r += 256) {
    const float c0 = coords[2 * r], c1 = coords[2 * r + 1];
    float a0 = 0.f, a1 = 0.f, a2 = 0.f;
    for (int j = 0; j < 512; ++j) {
      float h = fmaxf(c0 * cpb_w1[2 * j] + c1 * cpb_w1[2 * j + 1] + cpb_b1[j], 0.f);
      a0 += h * cpb_w2[0 * 512 + j];
      a1 += h * cpb_w2[1 * 512 + j];
      a2 += h * cpb_w2[2 * 512 + j];
    }
    tbl[r * NH + 0] = a0; tbl[r * NH + 1] = a1; tbl[r * NH + 2] = a2;
  }
  __syncthreads();

  // bias16[h][n][m] = 16 * sigmoid(tbl[rel_index[n][m]][h])
  for (int i = tid; i < NH * N_TOK * N_TOK; i += 256) {
    const int h  = i >> 12;
    const int nm = i & 4095;
    const float v = tbl[rel_index[nm] * NH + h];
    bias16[i] = 16.0f / (1.0f + __expf(-v));
  }
}

// ---------------------------------------------------------------------------
// Fused window attention: one workgroup (8 wave32) per window.
// ---------------------------------------------------------------------------
__global__ __launch_bounds__(256) void swin_attn_kernel(
    const float* __restrict__ x, const float* __restrict__ mask,
    const float* __restrict__ logit_scale, const float* __restrict__ proj_b,
    float* __restrict__ out, const void* __restrict__ wsv) {
  const __bf16* w_qkv  = (const __bf16*)((const char*)wsv + WS_WQKV);
  const __bf16* w_proj = (const __bf16*)((const char*)wsv + WS_WPROJ);
  const float*  qkvb   = (const float*)((const char*)wsv + WS_QKVB);
  const float*  bias16 = (const float*)((const char*)wsv + WS_BIAS16);

  // Dynamic LDS carve-up (120 KB total; s_qkv region reused for attn/attnb).
  extern __shared__ char smem[];
  __bf16* s_xo  = (__bf16*)(smem);           // 12288 B: x bf16, later o bf16
  float*  s_qkv = (float*)(smem + 12288);    // 73728 B: qkv f32 / attn / attnb
  __bf16* s_qn  = (__bf16*)(smem + 86016);   // 12288 B
  __bf16* s_kn  = (__bf16*)(smem + 98304);   // 12288 B
  __bf16* s_vt  = (__bf16*)(smem + 110592);  // 12288 B  [ch][tok]

  const int tid  = threadIdx.x;
  const int lane = tid & 31;
  // Wave id is uniform across the wave: force it scalar so every tile loop
  // below is wave-uniform (no EXEC masking around WMMA).
  const int wv   = __builtin_amdgcn_readfirstlane(tid >> 5);
  const float* xw = x + (size_t)blockIdx.x * (N_TOK * C_DIM);

  // ---- Phase 1: stage x as bf16 -------------------------------------------
  __builtin_prefetch(xw, 0, 0);  // global_prefetch_b8 warm-up
  for (int i = tid; i < N_TOK * C_DIM; i += 256) s_xo[i] = (__bf16)xw[i];
  __syncthreads();

  // ---- Phase 1b: qkv = x @ Wqkv^T + b   (64x96)x(96x288) ------------------
  // 72 tiles (4 M x 18 N), exactly 9 per wave.
  for (int it = 0; it < 9; ++it) {
    const int t  = wv + (it << 3);
    const int m0 = (t & 3) << 4;
    const int n0 = (t >> 2) << 4;
    v8f acc = {};
#pragma unroll
    for (int k0 = 0; k0 < C_DIM; k0 += 32) {
      v16bf a = frag_ld(s_xo, C_DIM, m0, k0, lane);
      v16bf b = frag_ld(w_qkv, C_DIM, n0, k0, lane);  // W row-major = B^T slice
      acc = wmma_bf16(a, b, acc);
    }
    const int n  = n0 + (lane & 15);
    const int mb = m0 + ((lane >> 4) << 3);
    const float bb = qkvb[n];
#pragma unroll
    for (int r = 0; r < 8; ++r) s_qkv[(mb + r) * QKV_OUT + n] = acc[r] + bb;
  }
  __syncthreads();

  // ---- Phase 2: cosine-normalize q,k (scale folded into q); transpose v ---
  for (int task = tid; task < 384; task += 256) {
    const int tok = task & 63;
    const int h   = (task >> 6) % NH;
    const int isK = task / 192;
    const float* src = s_qkv + tok * QKV_OUT + isK * C_DIM + h * 32;
    float ss = 0.f;
#pragma unroll
    for (int d = 0; d < 32; ++d) { const float v = src[d]; ss += v * v; }
    float inv = 1.0f / fmaxf(sqrtf(ss), 1e-12f);
    if (!isK) inv *= __expf(fminf(logit_scale[h], 4.6051702f));  // ln(100)
    __bf16* dst = (isK ? s_kn : s_qn) + tok * C_DIM + h * 32;
#pragma unroll
    for (int d = 0; d < 32; ++d) dst[d] = (__bf16)(src[d] * inv);
  }
  for (int i = tid; i < N_TOK * C_DIM; i += 256) {
    const int tok = i / C_DIM, ch = i % C_DIM;
    s_vt[ch * N_TOK + tok] = (__bf16)s_qkv[tok * QKV_OUT + 2 * C_DIM + ch];
  }
  __syncthreads();

  // s_qkv region now reused: attn f32 [3][64][64] + attnb bf16 [3][64][64]
  float*  s_attn  = s_qkv;
  __bf16* s_attnb = (__bf16*)(s_qkv + NH * N_TOK * N_TOK);

  // ---- Phase 3: attn = qn @ kn^T  (48 tiles, exactly 6 per wave) ----------
  for (int it = 0; it < 6; ++it) {
    const int t   = wv + (it << 3);
    const int h   = t >> 4;
    const int rem = t & 15;
    const int m0  = (rem & 3) << 4;
    const int n0  = (rem >> 2) << 4;
    v16bf a = frag_ld(s_qn, C_DIM, m0, h * 32, lane);
    v16bf b = frag_ld(s_kn, C_DIM, n0, h * 32, lane);
    v8f acc = {};
    acc = wmma_bf16(a, b, acc);
    const int n  = n0 + (lane & 15);
    const int mb = m0 + ((lane >> 4) << 3);
    float* dsth = s_attn + h * 4096;
#pragma unroll
    for (int r = 0; r < 8; ++r) dsth[(mb + r) * N_TOK + n] = acc[r];
  }
  __syncthreads();

  // ---- Phase 3b: + cpb bias + shift mask, softmax, -> bf16 probs ----------
  const int wIdx = blockIdx.x & (NW - 1);
  for (int r = tid; r < NH * N_TOK; r += 256) {
    const int h = r >> 6, tok = r & 63;
    float* row      = s_attn + (h * N_TOK + tok) * N_TOK;
    const float* bp = bias16 + (h * N_TOK + tok) * N_TOK;
    const float* mp = mask + (size_t)wIdx * 4096 + tok * N_TOK;
    float mx = -1e30f;
    for (int m = 0; m < N_TOK; ++m) mx = fmaxf(mx, row[m] + bp[m] + mp[m]);
    float sum = 0.f;
    __bf16* ob = s_attnb + (h * N_TOK + tok) * N_TOK;
    for (int m = 0; m < N_TOK; ++m) {
      const float e = __expf(row[m] + bp[m] + mp[m] - mx);
      ob[m] = (__bf16)e;
      sum += e;
    }
    const float inv = 1.0f / sum;
    for (int m = 0; m < N_TOK; ++m) ob[m] = (__bf16)((float)ob[m] * inv);
  }
  __syncthreads();

  // ---- Phase 4: out_h = attn @ v  (24 tiles, exactly 3 per wave) ----------
  __bf16* s_o = s_xo;  // reuse x staging buffer
  for (int it = 0; it < 3; ++it) {
    const int t   = wv + (it << 3);
    const int h   = t >> 3;
    const int rem = t & 7;
    const int m0  = (rem & 3) << 4;
    const int n0  = (rem >> 2) << 4;
    v8f acc = {};
#pragma unroll
    for (int k0 = 0; k0 < N_TOK; k0 += 32) {
      v16bf a = frag_ld(s_attnb + h * 4096, N_TOK, m0, k0, lane);
      v16bf b = frag_ld(s_vt + (h * 32) * N_TOK, N_TOK, n0, k0, lane);
      acc = wmma_bf16(a, b, acc);
    }
    const int n  = n0 + (lane & 15);
    const int mb = m0 + ((lane >> 4) << 3);
#pragma unroll
    for (int r = 0; r < 8; ++r)
      s_o[(mb + r) * C_DIM + h * 32 + n] = (__bf16)acc[r];
  }
  __syncthreads();

  // ---- Phase 5: out = o @ Wproj^T + b  (24 tiles, exactly 3 per wave) -----
  float* ow = out + (size_t)blockIdx.x * (N_TOK * C_DIM);
  for (int it = 0; it < 3; ++it) {
    const int t  = wv + (it << 3);
    const int m0 = (t & 3) << 4;
    const int n0 = (t >> 2) << 4;
    v8f acc = {};
#pragma unroll
    for (int k0 = 0; k0 < C_DIM; k0 += 32) {
      v16bf a = frag_ld(s_o, C_DIM, m0, k0, lane);
      v16bf b = frag_ld(w_proj, C_DIM, n0, k0, lane);
      acc = wmma_bf16(a, b, acc);
    }
    const int n  = n0 + (lane & 15);
    const int mb = m0 + ((lane >> 4) << 3);
    const float pb = proj_b[n];
#pragma unroll
    for (int r = 0; r < 8; ++r) ow[(mb + r) * C_DIM + n] = acc[r] + pb;
  }
}

extern "C" void kernel_launch(void* const* d_in, const int* in_sizes, int n_in,
                              void* d_out, int out_size, void* d_ws, size_t ws_size,
                              hipStream_t stream) {
  const float* x      = (const float*)d_in[0];
  const float* mask   = (const float*)d_in[1];
  const float* qkv_w  = (const float*)d_in[2];
  const float* q_bias = (const float*)d_in[3];
  const float* v_bias = (const float*)d_in[4];
  const float* lscale = (const float*)d_in[5];
  const float* cpb_w1 = (const float*)d_in[6];
  const float* cpb_b1 = (const float*)d_in[7];
  const float* cpb_w2 = (const float*)d_in[8];
  const float* proj_w = (const float*)d_in[9];
  const float* proj_b = (const float*)d_in[10];
  const float* coords = (const float*)d_in[11];
  const int*   relidx = (const int*)d_in[12];
  float* out = (float*)d_out;
  (void)in_sizes; (void)n_in; (void)out_size; (void)ws_size;

  swin_prep_kernel<<<1, 256, 0, stream>>>(qkv_w, proj_w, q_bias, v_bias,
                                          cpb_w1, cpb_b1, cpb_w2, coords,
                                          relidx, d_ws);
  const int nblocks = 8192;        // B_IMG * nW windows
  const size_t lds_bytes = 122880; // 120 KB dynamic LDS (2 WGs fit per 320KB WGP)
  swin_attn_kernel<<<nblocks, 256, lds_bytes, stream>>>(x, mask, lscale,
                                                        proj_b, out, d_ws);
}
